// MultiHeadAttention_3496103379423
// MI455X (gfx1250) — matrix operand order repaired and checked by reading, not compiled, not run on hardware
//
#include <hip/hip_runtime.h>

typedef __bf16 bf16_t;
typedef __attribute__((ext_vector_type(16))) __bf16 v16bf;
typedef __attribute__((ext_vector_type(8)))  __bf16 v8bf;
typedef __attribute__((ext_vector_type(4)))  __bf16 v4bf;
typedef __attribute__((ext_vector_type(8)))  float  v8f;
typedef __attribute__((ext_vector_type(4)))  float  v4f;
typedef __attribute__((ext_vector_type(4)))  unsigned int u32x4;
typedef __attribute__((ext_vector_type(8)))  int i32x8;

#define B_DIM  2
#define S_DIM  4096
#define D_DIM  512
#define H_DIM  8
#define HD_DIM 64

static __device__ inline v16bf cat8(v8bf lo, v8bf hi) {
  v16bf r;
#pragma unroll
  for (int i = 0; i < 8; ++i) { r[i] = lo[i]; r[i + 8] = hi[i]; }
  return r;
}

static __device__ inline v8f wmma_bf16(v16bf a, v16bf b, v8f c) {
  // D = A(16x32 bf16) x B(32x16 bf16) + C(16x16 f32)
  return __builtin_amdgcn_wmma_f32_16x16x32_bf16(false, a, false, b,
                                                 (short)0, c, false, false);
}

// A-matrix fragment (16x32 bf16) from row-major [m][k] bf16 (stride in halves).
static __device__ inline v16bf load_a_frag(const bf16_t* base, int stride, int lane) {
  const bf16_t* p = base + (lane & 15) * stride + ((lane < 16) ? 0 : 8);
  return cat8(*(const v8bf*)(p), *(const v8bf*)(p + 16));
}

// B-matrix fragment (32x16 bf16) from n-major [n][k] bf16 (stride in halves).
static __device__ inline v16bf load_b_frag(const bf16_t* baseN, int stride, int lane) {
  const bf16_t* p = baseN + (lane & 15) * stride + ((lane < 16) ? 0 : 8);
  return cat8(*(const v8bf*)(p), *(const v8bf*)(p + 16));
}

// ---------------------------------------------------------------------------
// Tensor Data Mover: DMA a 2D bf16 tile (rows x width, row stride in elems)
// from global memory into LDS at lds_off, with optional LDS row padding.
// D# built per CDNA5 ISA 8.3/8.4 (group0: count/lds/global/type,
// group1: data_size, pad, dims, tile, strides).  2-SGPR-group form.
// ---------------------------------------------------------------------------
static __device__ inline void tdm_load_2d(unsigned lds_off, const bf16_t* gaddr,
                                          unsigned rows, unsigned width,
                                          unsigned row_stride,
                                          unsigned pad_en, unsigned pad_int_code,
                                          unsigned pad_amt_code) {
  unsigned long long ga = (unsigned long long)(size_t)gaddr;
  u32x4 g0;
  g0[0] = 1u;                                          // count=1, user desc
  g0[1] = lds_off;                                     // lds_addr (bytes)
  g0[2] = (unsigned)(ga & 0xffffffffu);                // global_addr[31:0]
  g0[3] = (unsigned)((ga >> 32) & 0x01ffffffu)         // global_addr[56:32]
        | (2u << 30);                                  // type = 2 ("image")
  i32x8 g1;
  g1[0] = (int)((1u << 16)                             // data_size = 2 bytes
        | (pad_en << 20)                               // pad_enable
        | (pad_int_code << 22)                         // pad_interval
        | (pad_amt_code << 25));                       // pad_amount
  g1[1] = (int)(width << 16);                          // tensor_dim0[15:0] @ b63:48
  g1[2] = (int)(rows << 16);                           // tensor_dim1[15:0] @ b95:80
  g1[3] = (int)(width << 16);                          // tile_dim0 @ b127:112
  g1[4] = (int)(rows & 0xffffu);                       // tile_dim1 @ b143:128
  g1[5] = (int)row_stride;                             // tensor_dim0_stride[31:0]
  g1[6] = 0;
  g1[7] = 0;
  asm volatile("tensor_load_to_lds %0, %1" :: "s"(g0), "s"(g1) : "memory");
}

// ---------------------------------------------------------------------------
// GEMM + bias:  Out[M,N] = A[M,K] @ W[K,N] + bias[N]
// ---------------------------------------------------------------------------
template <bool A_F32, bool OUT_F32>
__global__ void __launch_bounds__(256)
gemm_bias_kernel(const void* __restrict__ Ain, const float* __restrict__ W,
                 const float* __restrict__ bias, void* __restrict__ Out,
                 int M, int K, int N) {
  const int ASTR = 40;
  __shared__ bf16_t sA[128 * 40];   // [m][k] tile, bf16
  __shared__ bf16_t sB[128 * 40];   // [n][k] tile (transposed W), bf16

  const int m0 = blockIdx.x * 128;
  const int n0 = blockIdx.y * 128;
  const int t = threadIdx.x;
  const int wave = t >> 5;
  const int lane = t & 31;
  const int mo = (wave >> 1) * 32;
  const int no = (wave & 1) * 64;

  v8f acc[2][4];
#pragma unroll
  for (int i = 0; i < 2; ++i)
#pragma unroll
    for (int j = 0; j < 4; ++j)
#pragma unroll
      for (int r = 0; r < 8; ++r) acc[i][j][r] = 0.0f;

  for (int k0 = 0; k0 < K; k0 += 32) {
#pragma unroll
    for (int i = 0; i < 4; ++i) {
      int c = t + i * 256;
      int e = c * 4;
      int row = e >> 5;
      int kk = e & 31;
      if (A_F32) {
        const float* A = (const float*)Ain;
        v4f x = *(const v4f*)(A + (size_t)(m0 + row) * K + k0 + kk);
        v4bf y;
#pragma unroll
        for (int j = 0; j < 4; ++j) y[j] = (bf16_t)x[j];
        *(v4bf*)(&sA[row * ASTR + kk]) = y;
      } else {
        const bf16_t* A = (const bf16_t*)Ain;
        *(v4bf*)(&sA[row * ASTR + kk]) =
            *(const v4bf*)(A + (size_t)(m0 + row) * K + k0 + kk);
      }
    }
#pragma unroll
    for (int i = 0; i < 4; ++i) {
      int c = t + i * 256;
      int e = c * 4;
      int kk = e >> 7;
      int n = e & 127;
      v4f x = *(const v4f*)(W + (size_t)(k0 + kk) * N + n0 + n);
#pragma unroll
      for (int j = 0; j < 4; ++j) sB[(n + j) * ASTR + kk] = (bf16_t)x[j];
    }
    if (k0 + 32 < K) {
      __builtin_prefetch((const float*)W + (size_t)(k0 + 32) * N + n0, 0, 1);
    }
    __syncthreads();

    v16bf af[2];
#pragma unroll
    for (int i = 0; i < 2; ++i)
      af[i] = load_a_frag(&sA[(mo + i * 16) * ASTR], ASTR, lane);
    v16bf bfr[4];
#pragma unroll
    for (int j = 0; j < 4; ++j)
      bfr[j] = load_b_frag(&sB[(no + j * 16) * ASTR], ASTR, lane);
#pragma unroll
    for (int i = 0; i < 2; ++i)
#pragma unroll
      for (int j = 0; j < 4; ++j) acc[i][j] = wmma_bf16(af[i], bfr[j], acc[i][j]);
    __syncthreads();
  }

#pragma unroll
  for (int j = 0; j < 4; ++j) {
    const int n = n0 + no + j * 16 + (lane & 15);
    const float bv = bias[n];
#pragma unroll
    for (int i = 0; i < 2; ++i) {
#pragma unroll
      for (int r = 0; r < 8; ++r) {
        const int m = m0 + mo + i * 16 + ((lane < 16) ? r : r + 8);
        const float v = acc[i][j][r] + bv;
        if (OUT_F32)
          ((float*)Out)[(size_t)m * N + n] = v;
        else
          ((bf16_t*)Out)[(size_t)m * N + n] = (bf16_t)v;
      }
    }
  }
}

// ---------------------------------------------------------------------------
// Flash attention over bf16 Q/K/V ([B,S,D], head h at column h*64).
// Block: 128 threads (4 waves), 64 q-rows per block, k-tiles of 32.
// K tile staged via Tensor Data Mover (tensor_load_to_lds + s_wait_tensorcnt);
// V tile staged transposed by the block; P round-trips C->LDS->A layout.
// ---------------------------------------------------------------------------
__global__ void __launch_bounds__(128)
flash_attn_kernel(const bf16_t* __restrict__ qb, const bf16_t* __restrict__ kb,
                  const bf16_t* __restrict__ vb, bf16_t* __restrict__ ctx) {
  const int S = S_DIM, D = D_DIM;
  const int VSTR = 40;
  const int KSTR = 72;                 // 64 halves + TDM pad (4 dwords/row)
  __shared__ bf16_t sVt[64 * 40];      // V tile transposed: [n][k]
  __shared__ bf16_t sK[32 * 72];       // K tile row-major [j][d], TDM-padded
  __shared__ bf16_t sP[4][16 * 32];    // per-wave P round-trip buffer [m][k]

  const int b = blockIdx.z;
  const int h = blockIdx.y;
  const int q0 = blockIdx.x * 64;
  const int t = threadIdx.x;
  const int wave = t >> 5;
  const int lane = t & 31;

  const size_t baseoff = ((size_t)b * S) * D + (size_t)h * HD_DIM;
  const bf16_t* qp = qb + baseoff;
  const bf16_t* kp = kb + baseoff;
  const bf16_t* vp = vb + baseoff;
  const unsigned sK_off = (unsigned)(size_t)(void*)&sK[0];  // LDS byte offset

  // Q A-fragments (16x64 = 2 frags of 16x32), loaded once from global.
  const bf16_t* qrow = qp + (size_t)(q0 + wave * 16 + (lane & 15)) * D;
  v16bf aq[2];
#pragma unroll
  for (int f = 0; f < 2; ++f) {
    const int kk = f * 32 + ((lane < 16) ? 0 : 8);
    aq[f] = cat8(*(const v8bf*)(qrow + kk), *(const v8bf*)(qrow + kk + 16));
  }

  v8f o[4];
#pragma unroll
  for (int j = 0; j < 4; ++j)
#pragma unroll
    for (int r = 0; r < 8; ++r) o[j][r] = 0.0f;
  float mrow[8], lrow[8];
#pragma unroll
  for (int r = 0; r < 8; ++r) { mrow[r] = -3.0e38f; lrow[r] = 0.0f; }

  for (int kt = 0; kt < S; kt += 32) {
    // ---- TDM: DMA K tile (32 rows x 64 d, row stride D) into sK ----
    if (wave == 0) {
      // pad_interval code 4 => every 32 DWORDs (one 64-elem bf16 row),
      // pad_amount code 3  => +4 DWORDs  => LDS row stride 72 halves.
      tdm_load_2d(sK_off, kp + (size_t)kt * D, 32u, 64u, (unsigned)D,
                  1u, 4u, 3u);
      __builtin_amdgcn_s_wait_tensorcnt(0);
    }
    // ---- stage V tile 32x64 transposed into sVt[n][k] ----
#pragma unroll
    for (int i = 0; i < 2; ++i) {
      int c = t + i * 128;
      int kk = c >> 3;
      int n8 = (c & 7) * 8;
      v8bf x = *(const v8bf*)(vp + (size_t)(kt + kk) * D + n8);
#pragma unroll
      for (int j = 0; j < 8; ++j) sVt[(n8 + j) * VSTR + kk] = x[j];
    }
    if (kt + 32 < S) {
      __builtin_prefetch(vp + (size_t)(kt + 32) * D, 0, 1);
    }
    __syncthreads();

    // ---- S = (Q K^T) * 1/sqrt(64); K B-frags from LDS ----
    v8f sf[2];
#pragma unroll
    for (int js = 0; js < 2; ++js) {
      v8f s;
#pragma unroll
      for (int r = 0; r < 8; ++r) s[r] = 0.0f;
      const bf16_t* krow = &sK[(js * 16 + (lane & 15)) * KSTR];
#pragma unroll
      for (int f = 0; f < 2; ++f) {
        const int d0 = f * 32 + ((lane < 16) ? 0 : 8);
        v16bf bk = cat8(*(const v8bf*)(krow + d0), *(const v8bf*)(krow + d0 + 16));
        s = wmma_bf16(aq[f], bk, s);
      }
#pragma unroll
      for (int r = 0; r < 8; ++r) s[r] *= 0.125f;
      sf[js] = s;
    }

    // ---- online softmax (rows live in lane-halves; 16-lane reductions) ----
    float mnew[8];
#pragma unroll
    for (int r = 0; r < 8; ++r) {
      float v = fmaxf(sf[0][r], sf[1][r]);
      v = fmaxf(v, __shfl_xor(v, 1, 32));
      v = fmaxf(v, __shfl_xor(v, 2, 32));
      v = fmaxf(v, __shfl_xor(v, 4, 32));
      v = fmaxf(v, __shfl_xor(v, 8, 32));
      mnew[r] = fmaxf(mrow[r], v);
    }
    float corr[8];
#pragma unroll
    for (int r = 0; r < 8; ++r) corr[r] = __expf(mrow[r] - mnew[r]);
#pragma unroll
    for (int js = 0; js < 2; ++js)
#pragma unroll
      for (int r = 0; r < 8; ++r) sf[js][r] = __expf(sf[js][r] - mnew[r]);
#pragma unroll
    for (int r = 0; r < 8; ++r) {
      float v = sf[0][r] + sf[1][r];
      v += __shfl_xor(v, 1, 32);
      v += __shfl_xor(v, 2, 32);
      v += __shfl_xor(v, 4, 32);
      v += __shfl_xor(v, 8, 32);
      lrow[r] = lrow[r] * corr[r] + v;
      mrow[r] = mnew[r];
    }
#pragma unroll
    for (int j = 0; j < 4; ++j)
#pragma unroll
      for (int r = 0; r < 8; ++r) o[j][r] *= corr[r];

    // ---- P: C-layout -> LDS -> A-layout ----
    bf16_t* Pw = &sP[wave][0];
#pragma unroll
    for (int js = 0; js < 2; ++js)
#pragma unroll
      for (int r = 0; r < 8; ++r)
        Pw[((lane < 16) ? r : r + 8) * 32 + js * 16 + (lane & 15)] =
            (bf16_t)sf[js][r];
    const bf16_t* pr = Pw + (lane & 15) * 32 + ((lane < 16) ? 0 : 8);
    v16bf pa = cat8(*(const v8bf*)(pr), *(const v8bf*)(pr + 16));

    // ---- O += P V ----
#pragma unroll
    for (int j = 0; j < 4; ++j) {
      v16bf bv = load_b_frag(&sVt[(j * 16) * VSTR], VSTR, lane);
      o[j] = wmma_bf16(pa, bv, o[j]);
    }
    __syncthreads();
  }

  // ---- normalize and store ctx (bf16) ----
  float inv[8];
#pragma unroll
  for (int r = 0; r < 8; ++r) inv[r] = 1.0f / lrow[r];
#pragma unroll
  for (int j = 0; j < 4; ++j) {
#pragma unroll
    for (int r = 0; r < 8; ++r) {
      const int m = q0 + wave * 16 + ((lane < 16) ? r : r + 8);
      ctx[baseoff + (size_t)m * D + j * 16 + (lane & 15)] =
          (bf16_t)(o[j][r] * inv[r]);
    }
  }
}

// ---------------------------------------------------------------------------
extern "C" void kernel_launch(void* const* d_in, const int* in_sizes, int n_in,
                              void* d_out, int out_size, void* d_ws, size_t ws_size,
                              hipStream_t stream) {
  (void)in_sizes; (void)n_in; (void)out_size; (void)ws_size;
  const float* q  = (const float*)d_in[0];
  const float* k  = (const float*)d_in[1];
  const float* v  = (const float*)d_in[2];
  const float* Wq = (const float*)d_in[3];
  const float* bq = (const float*)d_in[4];
  const float* Wk = (const float*)d_in[5];
  const float* bk = (const float*)d_in[6];
  const float* Wv = (const float*)d_in[7];
  const float* bv = (const float*)d_in[8];
  const float* Wo = (const float*)d_in[9];
  const float* bo = (const float*)d_in[10];

  const int M = B_DIM * S_DIM;  // 8192
  const int K = D_DIM, N = D_DIM;
  const size_t elems = (size_t)B_DIM * S_DIM * D_DIM;

  bf16_t* qb  = (bf16_t*)d_ws;
  bf16_t* kbf = qb + elems;
  bf16_t* vbf = kbf + elems;
  bf16_t* ctx = vbf + elems;

  dim3 gg(M / 128, N / 128, 1);
  gemm_bias_kernel<true, false><<<gg, 256, 0, stream>>>(q, Wq, bq, qb, M, K, N);
  gemm_bias_kernel<true, false><<<gg, 256, 0, stream>>>(k, Wk, bk, kbf, M, K, N);
  gemm_bias_kernel<true, false><<<gg, 256, 0, stream>>>(v, Wv, bv, vbf, M, K, N);

  dim3 ga(S_DIM / 64, H_DIM, B_DIM);
  flash_attn_kernel<<<ga, 128, 0, stream>>>(qb, kbf, vbf, ctx);

  gemm_bias_kernel<false, true><<<gg, 256, 0, stream>>>(ctx, Wo, bo, d_out, M, K, N);
}
